// Combiner_28037546508565
// MI455X (gfx1250) — matrix operand order repaired and checked by reading, not compiled, not run on hardware
//
#include <hip/hip_runtime.h>

typedef __bf16 bf16;
typedef __bf16 v16bf __attribute__((ext_vector_type(16)));
typedef __bf16 v8bf  __attribute__((ext_vector_type(8)));
typedef float  v8f   __attribute__((ext_vector_type(8)));

#define DCC 512
#define LDA 40   // padded LDS row stride in elements (80B: conflict-free b128 frag reads)

__device__ __forceinline__ float sigm(float x)  { return 1.0f / (1.0f + __expf(-x)); }
__device__ __forceinline__ float tanh_(float x) { return 2.0f / (1.0f + __expf(-2.0f * x)) - 1.0f; }

__device__ __forceinline__ v8f splat8(float x) {
    v8f r;
#pragma unroll
    for (int j = 0; j < 8; ++j) r[j] = x;
    return r;
}

__device__ __forceinline__ v16bf ld_afrag(const bf16* p, int kb) {
    // A frag (16x32): lane<16 -> K {0..7,16..23}; lane>=16 -> K {8..15,24..31}
    v16bf a;
    ((uint4*)&a)[0] = *(const uint4*)(p + kb);
    ((uint4*)&a)[1] = *(const uint4*)(p + kb + 16);
    return a;
}
__device__ __forceinline__ v16bf ld_bfrag(const bf16* p, int kb) {
    // B frag (32x16): lane<16 -> K {0..7,16..23}; lane>=16 -> K {8..15,24..31}
    v16bf b;
    ((uint4*)&b)[0] = *(const uint4*)(p + kb);
    ((uint4*)&b)[1] = *(const uint4*)(p + kb + 16);
    return b;
}

#define WMMA(acc, a, b) \
    acc = __builtin_amdgcn_wmma_f32_16x16x32_bf16(false, a, false, b, (short)0, acc, false, false)

// ---------------- small prep kernels ----------------
__global__ void cvt_f32_bf16(const float* __restrict__ x, bf16* __restrict__ y, int n) {
    int i = blockIdx.x * blockDim.x + threadIdx.x;
    if (i < n) y[i] = (bf16)x[i];
}

__global__ void bias_sum_k(const float* __restrict__ a, const float* __restrict__ b,
                           float* __restrict__ o, int n) {
    int i = blockIdx.x * blockDim.x + threadIdx.x;
    if (i < n) o[i] = a[i] + b[i];
}

// ---------------- LSTM step 0 ----------------
// 64x64 h1/c1 tile per block; only gates i,g,o needed (f*c0 == 0).
__launch_bounds__(256)
__global__ void lstm_step0(const float* __restrict__ image_z, const float* __restrict__ text_z,
                           const int* __restrict__ img_m, const int* __restrict__ txt_m,
                           const bf16* __restrict__ wih, const float* __restrict__ bias,
                           bf16* __restrict__ h1, bf16* __restrict__ c1)
{
    const int AS_BUF = 64 * LDA;
    const int BS_BUF = 3 * 64 * LDA;
    __shared__ __align__(16) bf16 As[2 * 64 * LDA];
    __shared__ __align__(16) bf16 Bs[2 * 3 * 64 * LDA];   // gates i,g,o

    const int tid = threadIdx.x;
    const int n0 = blockIdx.x * 64;
    const int m0 = blockIdx.y * 64;

    const int lane = tid & 31;
    const int w    = tid >> 5;
    const int rb0  = (w >> 2) * 32;
    const int rb1  = rb0 + 16;
    const int cb   = (w & 3) * 16;
    const int lr   = lane & 15;
    const int hi   = lane >> 4;
    const int akb  = hi * 8;
    const int bkb  = hi * 8;

    // A loader: 8 consecutive k of one row per thread
    const int arow  = tid >> 2;
    const int akloc = (tid & 3) * 8;
    const int grow  = m0 + arow;
    const int img_ok = (img_m[grow] == 0);
    const int txt_ok = (txt_m[grow] == 0);
    const float* arow_p = img_ok ? (image_z + (size_t)grow * DCC)
                                 : (text_z + (size_t)grow * 256);

    // B loader: threads 0..191 each stage one (gate,col) 64B slice; gate map {i,g,o}->rows {0,2,3}
    const int bg = tid >> 6;                 // 0..3
    const int bc = tid & 63;
    const int wgate = (bg == 0) ? 0 : (bg + 1);
    const bf16* wrow = wih + (size_t)(wgate * DCC + n0 + bc) * DCC;

    bf16* As_st = &As[arow * LDA + akloc];
    bf16* Bs_st = &Bs[(bg * 64 + bc) * LDA];
    const bf16* Af0 = &As[(rb0 + lr) * LDA];
    const bf16* Af1 = &As[(rb1 + lr) * LDA];
    const bf16* BfI = &Bs[(0 * 64 + cb + lr) * LDA];
    const bf16* BfG = &Bs[(1 * 64 + cb + lr) * LDA];
    const bf16* BfO = &Bs[(2 * 64 + cb + lr) * LDA];

    const int col = n0 + cb + lr;
    const float bi  = bias[col];
    const float bg_ = bias[2 * DCC + col];
    const float bo  = bias[3 * DCC + col];
    v8f aI0 = splat8(bi), aG0 = splat8(bg_), aO0 = splat8(bo);
    v8f aI1 = aI0, aG1 = aG0, aO1 = aO0;

    auto prefA = [&](int k0) -> v8bf {
        const int k = k0 + akloc;
        const int koff = img_ok ? k : (k & 255);            // wrap keeps text reads in-bounds
        const bool valid = img_ok || (txt_ok && k < 256);
        float4 u = *(const float4*)(arow_p + koff);
        float4 vv = *(const float4*)(arow_p + koff + 4);
        v8bf r;
        r[0] = (bf16)(valid ? u.x : 0.f);  r[1] = (bf16)(valid ? u.y : 0.f);
        r[2] = (bf16)(valid ? u.z : 0.f);  r[3] = (bf16)(valid ? u.w : 0.f);
        r[4] = (bf16)(valid ? vv.x : 0.f); r[5] = (bf16)(valid ? vv.y : 0.f);
        r[6] = (bf16)(valid ? vv.z : 0.f); r[7] = (bf16)(valid ? vv.w : 0.f);
        return r;
    };

    v8bf areg; uint4 br0, br1, br2, br3;
    auto pref = [&](int k0) {
        areg = prefA(k0);
        const uint4* s = (const uint4*)(wrow + k0);
        br0 = s[0]; br1 = s[1]; br2 = s[2]; br3 = s[3];
    };
    auto stAB = [&](int buf) {
        *(v8bf*)(As_st + buf * AS_BUF) = areg;
        if (bg < 3) {
            uint4* d = (uint4*)(Bs_st + buf * BS_BUF);
            d[0] = br0; d[1] = br1; d[2] = br2; d[3] = br3;
        }
    };
    auto compute = [&](int buf) {
        const int ao = buf * AS_BUF, bo2 = buf * BS_BUF;
        v16bf fa0 = ld_afrag(Af0 + ao, akb);
        v16bf fa1 = ld_afrag(Af1 + ao, akb);
        v16bf fbI = ld_bfrag(BfI + bo2, bkb);
        v16bf fbG = ld_bfrag(BfG + bo2, bkb);
        v16bf fbO = ld_bfrag(BfO + bo2, bkb);
        WMMA(aI0, fa0, fbI); WMMA(aG0, fa0, fbG); WMMA(aO0, fa0, fbO);
        WMMA(aI1, fa1, fbI); WMMA(aG1, fa1, fbG); WMMA(aO1, fa1, fbO);
    };

    pref(0);
    stAB(0);
    __syncthreads();

    for (int k0 = 0; k0 < DCC; k0 += 64) {
        pref(k0 + 32);
        compute(0);
        stAB(1);
        __syncthreads();

        const int kn = (k0 + 64 < DCC) ? k0 + 64 : 0;   // clamped (last store is dead)
        pref(kn);
        compute(1);
        stAB(0);
        __syncthreads();
    }

    auto epi = [&](const v8f& AI, const v8f& AG, const v8f& AO, int rowbase) {
#pragma unroll
        for (int j = 0; j < 8; ++j) {
            const int row = rowbase + j;
            const float iv = sigm(AI[j]);
            const float gv = tanh_(AG[j]);
            const float ov = sigm(AO[j]);
            const float cv = iv * gv;
            const float hv = ov * tanh_(cv);
            h1[(size_t)row * DCC + col] = (bf16)hv;
            c1[(size_t)row * DCC + col] = (bf16)cv;
        }
    };
    epi(aI0, aG0, aO0, m0 + rb0 + hi * 8);
    epi(aI1, aG1, aO1, m0 + rb1 + hi * 8);
}

// ---------------- LSTM step 1 ----------------
// gates1 = X1 @ Wih[:, :256]^T + h1 @ Whh^T + bias ; h_final = both ? h2 : h1
__launch_bounds__(256)
__global__ void lstm_step1(const float* __restrict__ text_z,
                           const int* __restrict__ img_m, const int* __restrict__ txt_m,
                           const bf16* __restrict__ wih, const bf16* __restrict__ whh,
                           const float* __restrict__ bias,
                           const bf16* __restrict__ h1, const bf16* __restrict__ c1,
                           bf16* __restrict__ hf)
{
    const int AS_BUF = 64 * LDA;
    const int BS_BUF = 4 * 64 * LDA;
    __shared__ __align__(16) bf16 As[2 * 64 * LDA];
    __shared__ __align__(16) bf16 Bs[2 * 4 * 64 * LDA];

    const int tid = threadIdx.x;
    const int n0 = blockIdx.x * 64;
    const int m0 = blockIdx.y * 64;

    const int lane = tid & 31;
    const int w    = tid >> 5;
    const int rb0  = (w >> 2) * 32;
    const int rb1  = rb0 + 16;
    const int cb   = (w & 3) * 16;
    const int lr   = lane & 15;
    const int hi   = lane >> 4;
    const int akb  = hi * 8;
    const int bkb  = hi * 8;

    const int arow  = tid >> 2;
    const int akloc = (tid & 3) * 8;
    const int grow  = m0 + arow;
    const int both_ld = (img_m[grow] == 0) && (txt_m[grow] == 0);

    const int bg = tid >> 6;
    const int bc = tid & 63;
    const bf16* wrow_ih = wih + (size_t)(bg * DCC + n0 + bc) * DCC;
    const bf16* wrow_hh = whh + (size_t)(bg * DCC + n0 + bc) * DCC;

    bf16* As_st = &As[arow * LDA + akloc];
    bf16* Bs_st = &Bs[(bg * 64 + bc) * LDA];
    const bf16* Af0 = &As[(rb0 + lr) * LDA];
    const bf16* Af1 = &As[(rb1 + lr) * LDA];
    const bf16* Bf0 = &Bs[(0 * 64 + cb + lr) * LDA];
    const bf16* Bf1 = &Bs[(1 * 64 + cb + lr) * LDA];
    const bf16* Bf2 = &Bs[(2 * 64 + cb + lr) * LDA];
    const bf16* Bf3 = &Bs[(3 * 64 + cb + lr) * LDA];

    const int col = n0 + cb + lr;
    const float bi  = bias[col];
    const float bf_ = bias[DCC + col];
    const float bg_ = bias[2 * DCC + col];
    const float bo  = bias[3 * DCC + col];
    v8f aI0 = splat8(bi), aF0 = splat8(bf_), aG0 = splat8(bg_), aO0 = splat8(bo);
    v8f aI1 = aI0, aF1 = aF0, aG1 = aG0, aO1 = aO0;

    v8bf areg; uint4 br0, br1, br2, br3;
    // unified prefetch over 24 K-steps: kk<8 -> (X1, Wih[:, :256]); kk>=8 -> (h1, Whh)
    auto pref = [&](int kk) {
        if (kk < 8) {
            const int k = kk * 32 + akloc;
            float4 u  = *(const float4*)(text_z + (size_t)grow * 256 + k);
            float4 vv = *(const float4*)(text_z + (size_t)grow * 256 + k + 4);
            areg[0] = (bf16)(both_ld ? u.x : 0.f);  areg[1] = (bf16)(both_ld ? u.y : 0.f);
            areg[2] = (bf16)(both_ld ? u.z : 0.f);  areg[3] = (bf16)(both_ld ? u.w : 0.f);
            areg[4] = (bf16)(both_ld ? vv.x : 0.f); areg[5] = (bf16)(both_ld ? vv.y : 0.f);
            areg[6] = (bf16)(both_ld ? vv.z : 0.f); areg[7] = (bf16)(both_ld ? vv.w : 0.f);
            const uint4* s = (const uint4*)(wrow_ih + kk * 32);
            br0 = s[0]; br1 = s[1]; br2 = s[2]; br3 = s[3];
        } else {
            const int k = (kk - 8) * 32 + akloc;
            areg = *(const v8bf*)(h1 + (size_t)grow * DCC + k);
            const uint4* s = (const uint4*)(wrow_hh + (kk - 8) * 32);
            br0 = s[0]; br1 = s[1]; br2 = s[2]; br3 = s[3];
        }
    };
    auto stAB = [&](int buf) {
        *(v8bf*)(As_st + buf * AS_BUF) = areg;
        uint4* d = (uint4*)(Bs_st + buf * BS_BUF);
        d[0] = br0; d[1] = br1; d[2] = br2; d[3] = br3;
    };
    auto compute = [&](int buf) {
        const int ao = buf * AS_BUF, bo2 = buf * BS_BUF;
        v16bf fa0 = ld_afrag(Af0 + ao, akb);
        v16bf fa1 = ld_afrag(Af1 + ao, akb);
        v16bf fb0 = ld_bfrag(Bf0 + bo2, bkb);
        v16bf fb1 = ld_bfrag(Bf1 + bo2, bkb);
        v16bf fb2 = ld_bfrag(Bf2 + bo2, bkb);
        v16bf fb3 = ld_bfrag(Bf3 + bo2, bkb);
        WMMA(aI0, fa0, fb0); WMMA(aF0, fa0, fb1); WMMA(aG0, fa0, fb2); WMMA(aO0, fa0, fb3);
        WMMA(aI1, fa1, fb0); WMMA(aF1, fa1, fb1); WMMA(aG1, fa1, fb2); WMMA(aO1, fa1, fb3);
    };

    pref(0);
    stAB(0);
    __syncthreads();

    for (int kk = 0; kk < 24; kk += 2) {
        pref(kk + 1);
        compute(0);
        stAB(1);
        __syncthreads();

        const int kn = (kk + 2 < 24) ? kk + 2 : 0;
        pref(kn);
        compute(1);
        stAB(0);
        __syncthreads();
    }

    auto epi = [&](const v8f& AI, const v8f& AF, const v8f& AG, const v8f& AO, int rowbase) {
#pragma unroll
        for (int j = 0; j < 8; ++j) {
            const int row = rowbase + j;
            const int bothr = (img_m[row] == 0) && (txt_m[row] == 0);
            const float iv = sigm(AI[j]);
            const float fv = sigm(AF[j]);
            const float gv = tanh_(AG[j]);
            const float ov = sigm(AO[j]);
            const float c1v = (float)c1[(size_t)row * DCC + col];
            const float cv = fv * c1v + iv * gv;
            const float h2 = ov * tanh_(cv);
            const bf16 h1v = h1[(size_t)row * DCC + col];
            hf[(size_t)row * DCC + col] = bothr ? (bf16)h2 : h1v;
        }
    };
    epi(aI0, aF0, aG0, aO0, m0 + rb0 + hi * 8);
    epi(aI1, aF1, aG1, aO1, m0 + rb1 + hi * 8);
}

// ---------------- FC head: out = hf @ Wfc^T + b_fc  (N = 1024) ----------------
__launch_bounds__(256)
__global__ void fc_kernel(const bf16* __restrict__ hf, const bf16* __restrict__ wfc,
                          const float* __restrict__ b_fc, float* __restrict__ out)
{
    const int AS_BUF = 64 * LDA;
    const int BS_BUF = 256 * LDA;
    __shared__ __align__(16) bf16 As[2 * 64 * LDA];
    __shared__ __align__(16) bf16 Bs[2 * 256 * LDA];

    const int tid = threadIdx.x;
    const int n0 = blockIdx.x * 256;
    const int m0 = blockIdx.y * 64;

    const int lane = tid & 31;
    const int w    = tid >> 5;
    const int rb0  = (w >> 2) * 32;
    const int rb1  = rb0 + 16;
    const int cb   = (w & 3) * 64;      // 4 subtiles of 16 cols per wave
    const int lr   = lane & 15;
    const int hi   = lane >> 4;
    const int akb  = hi * 8;
    const int bkb  = hi * 8;

    const int arow  = tid >> 2;
    const int akloc = (tid & 3) * 8;
    const int grow  = m0 + arow;

    const bf16* wrow = wfc + (size_t)(n0 + tid) * DCC;   // one output col per thread

    bf16* As_st = &As[arow * LDA + akloc];
    bf16* Bs_st = &Bs[tid * LDA];
    const bf16* Af0 = &As[(rb0 + lr) * LDA];
    const bf16* Af1 = &As[(rb1 + lr) * LDA];
    const bf16* Bq0 = &Bs[(cb +  0 + lr) * LDA];
    const bf16* Bq1 = &Bs[(cb + 16 + lr) * LDA];
    const bf16* Bq2 = &Bs[(cb + 32 + lr) * LDA];
    const bf16* Bq3 = &Bs[(cb + 48 + lr) * LDA];

    const int c0g = n0 + cb + lr;
    v8f a00 = splat8(b_fc[c0g]);
    v8f a01 = splat8(b_fc[c0g + 16]);
    v8f a02 = splat8(b_fc[c0g + 32]);
    v8f a03 = splat8(b_fc[c0g + 48]);
    v8f a10 = a00, a11 = a01, a12 = a02, a13 = a03;

    v8bf areg; uint4 br0, br1, br2, br3;
    auto pref = [&](int k0) {
        areg = *(const v8bf*)(hf + (size_t)grow * DCC + k0 + akloc);
        const uint4* s = (const uint4*)(wrow + k0);
        br0 = s[0]; br1 = s[1]; br2 = s[2]; br3 = s[3];
    };
    auto stAB = [&](int buf) {
        *(v8bf*)(As_st + buf * AS_BUF) = areg;
        uint4* d = (uint4*)(Bs_st + buf * BS_BUF);
        d[0] = br0; d[1] = br1; d[2] = br2; d[3] = br3;
    };
    auto compute = [&](int buf) {
        const int ao = buf * AS_BUF, bo2 = buf * BS_BUF;
        v16bf fa0 = ld_afrag(Af0 + ao, akb);
        v16bf fa1 = ld_afrag(Af1 + ao, akb);
        v16bf fb0 = ld_bfrag(Bq0 + bo2, bkb);
        v16bf fb1 = ld_bfrag(Bq1 + bo2, bkb);
        v16bf fb2 = ld_bfrag(Bq2 + bo2, bkb);
        v16bf fb3 = ld_bfrag(Bq3 + bo2, bkb);
        WMMA(a00, fa0, fb0); WMMA(a01, fa0, fb1); WMMA(a02, fa0, fb2); WMMA(a03, fa0, fb3);
        WMMA(a10, fa1, fb0); WMMA(a11, fa1, fb1); WMMA(a12, fa1, fb2); WMMA(a13, fa1, fb3);
    };

    pref(0);
    stAB(0);
    __syncthreads();

    for (int k0 = 0; k0 < DCC; k0 += 64) {
        pref(k0 + 32);
        compute(0);
        stAB(1);
        __syncthreads();

        const int kn = (k0 + 64 < DCC) ? k0 + 64 : 0;
        pref(kn);
        compute(1);
        stAB(0);
        __syncthreads();
    }

    auto epi = [&](const v8f& A0, const v8f& A1, const v8f& A2, const v8f& A3, int rowbase) {
#pragma unroll
        for (int j = 0; j < 8; ++j) {
            const size_t row = (size_t)(rowbase + j);
            __builtin_nontemporal_store(A0[j], &out[row * 1024 + c0g +  0]);
            __builtin_nontemporal_store(A1[j], &out[row * 1024 + c0g + 16]);
            __builtin_nontemporal_store(A2[j], &out[row * 1024 + c0g + 32]);
            __builtin_nontemporal_store(A3[j], &out[row * 1024 + c0g + 48]);
        }
    };
    epi(a00, a01, a02, a03, m0 + rb0 + hi * 8);
    epi(a10, a11, a12, a13, m0 + rb1 + hi * 8);
}

// ---------------- host launcher ----------------
extern "C" void kernel_launch(void* const* d_in, const int* in_sizes, int n_in,
                              void* d_out, int out_size, void* d_ws, size_t ws_size,
                              hipStream_t stream) {
    const float* image_z = (const float*)d_in[0];
    const float* text_z  = (const float*)d_in[1];
    const int*   img_m   = (const int*)d_in[2];
    const int*   txt_m   = (const int*)d_in[3];
    const float* W_ih    = (const float*)d_in[4];
    const float* W_hh    = (const float*)d_in[5];
    const float* b_ih    = (const float*)d_in[6];
    const float* b_hh    = (const float*)d_in[7];
    const float* W_fc    = (const float*)d_in[8];
    const float* b_fc    = (const float*)d_in[9];
    float* out = (float*)d_out;

    const int Bn = in_sizes[2];            // batch size (mask element count)

    char* ws = (char*)d_ws;
    const size_t OFF_WIH  = 0;                          // 2048*512 bf16 = 2 MB
    const size_t OFF_WHH  = OFF_WIH + 2048ull*512*2;    // 2 MB
    const size_t OFF_WFC  = OFF_WHH + 2048ull*512*2;    // 1 MB
    const size_t OFF_BIAS = OFF_WFC + 1024ull*512*2;    // 8 KB
    const size_t OFF_H1   = OFF_BIAS + 2048ull*4;
    const size_t OFF_C1   = OFF_H1 + (size_t)Bn*512*2;
    const size_t OFF_HF   = OFF_C1 + (size_t)Bn*512*2;

    bf16*  wih_bf = (bf16*)(ws + OFF_WIH);
    bf16*  whh_bf = (bf16*)(ws + OFF_WHH);
    bf16*  wfc_bf = (bf16*)(ws + OFF_WFC);
    float* bias   = (float*)(ws + OFF_BIAS);
    bf16*  h1     = (bf16*)(ws + OFF_H1);
    bf16*  c1     = (bf16*)(ws + OFF_C1);
    bf16*  hf     = (bf16*)(ws + OFF_HF);

    cvt_f32_bf16<<<(2048*512 + 255) / 256, 256, 0, stream>>>(W_ih, wih_bf, 2048 * 512);
    cvt_f32_bf16<<<(2048*512 + 255) / 256, 256, 0, stream>>>(W_hh, whh_bf, 2048 * 512);
    cvt_f32_bf16<<<(1024*512 + 255) / 256, 256, 0, stream>>>(W_fc, wfc_bf, 1024 * 512);
    bias_sum_k<<<(2048 + 255) / 256, 256, 0, stream>>>(b_ih, b_hh, bias, 2048);

    dim3 blk(256);
    // n-major grid: blocks sharing activation rows are adjacent in dispatch (L2-friendly)
    lstm_step0<<<dim3(512 / 64, Bn / 64), blk, 0, stream>>>(image_z, text_z, img_m, txt_m,
                                                            wih_bf, bias, h1, c1);
    lstm_step1<<<dim3(512 / 64, Bn / 64), blk, 0, stream>>>(text_z, img_m, txt_m,
                                                            wih_bf, whh_bf, bias, h1, c1, hf);
    fc_kernel<<<dim3(1024 / 256, Bn / 64), blk, 0, stream>>>(hf, wfc_bf, b_fc, out);
    (void)n_in; (void)out_size; (void)ws_size;
}
